// SelfAttention_9749575762112
// MI455X (gfx1250) — matrix operand order repaired and checked by reading, not compiled, not run on hardware
//
#include <hip/hip_runtime.h>
#include <stdint.h>

// ---------------------------------------------------------------------------
// Types for CDNA5 WMMA (wave32): v_wmma_f32_16x16x32_bf16
// ---------------------------------------------------------------------------
typedef __attribute__((ext_vector_type(16))) __bf16 v16bf;
typedef __attribute__((ext_vector_type(8)))  float  v8f;

static __device__ __forceinline__ v8f wmma_bf16(v16bf a, v16bf b, v8f c) {
  // 8 args: (neg_a, A, neg_b, B, c_mod, C, reuse_a, reuse_b)
  return __builtin_amdgcn_wmma_f32_16x16x32_bf16(false, a, false, b,
                                                 (short)0, c, false, false);
}

// float -> bf16 via hardware convert (RNE), raw u16 bit pattern
static __device__ __forceinline__ unsigned short f2bf(float f) {
  union { __bf16 h; unsigned short u; } c;
  c.h = (__bf16)f;
  return c.u;
}

// 16 contiguous bf16 (32 bytes) -> B-fragment half-pair
static __device__ __forceinline__ v16bf load_frag16(const unsigned short* p) {
  union { uint4 u[2]; v16bf v; } f;
  f.u[0] = *(const uint4*)(p);
  f.u[1] = *(const uint4*)(p + 8);
  return f.v;
}

// two separate 8-element (16 byte) runs -> A-fragment
static __device__ __forceinline__ v16bf load_frag_pair(const unsigned short* p0,
                                                       const unsigned short* p1) {
  union { uint4 u[2]; v16bf v; } f;
  f.u[0] = *(const uint4*)(p0);
  f.u[1] = *(const uint4*)(p1);
  return f.v;
}

// CDNA5 async copy: 16B global -> LDS per lane, tracked by ASYNCcnt.
// ldsOff is the wave-relative LDS byte offset (dynamic LDS starts at 0 here).
static __device__ __forceinline__ void async_g2lds_b128(unsigned ldsOff,
                                                        const void* gptr) {
  unsigned long long ga = (unsigned long long)(uintptr_t)gptr;
  asm volatile("global_load_async_to_lds_b128 %0, %1, off"
               :: "v"(ldsOff), "v"(ga) : "memory");
}
static __device__ __forceinline__ void wait_asynccnt0() {
  asm volatile("s_wait_asynccnt 0x0" ::: "memory");
}

// ---------------------------------------------------------------------------
// Problem constants
// ---------------------------------------------------------------------------
#define B_  4
#define S_  4096
#define D_  1024
#define A_  1024
#define BS_ (B_ * S_)   // 16384 rows of X

// ===========================================================================
// Kernel 1: QKV projection GEMM.
//   Q, K written row-major bf16 [b*S+s][a].
//   V written TRANSPOSED bf16:  Vt[b][a][s]  (so attention can async-copy
//   V tiles straight into B-fragment-friendly LDS layout).
//   Block tile 128(M) x 128(N), K-step 32, 8 waves, wave tile 64x32.
// ===========================================================================
__global__ __launch_bounds__(256) void qkv_gemm(
    const float* __restrict__ X,
    const float* __restrict__ Wq, const float* __restrict__ Wk,
    const float* __restrict__ Wv,
    unsigned short* __restrict__ Qb, unsigned short* __restrict__ Kb,
    unsigned short* __restrict__ Vt)
{
  const float* W;
  unsigned short* Out;
  if (blockIdx.z == 0)      { W = Wq; Out = Qb; }
  else if (blockIdx.z == 1) { W = Wk; Out = Kb; }
  else                      { W = Wv; Out = Vt; }

  __shared__ __align__(16) unsigned short Xs[128 * 32];   // [m][k] bf16
  __shared__ __align__(16) unsigned short Ws[128 * 32];   // [n][k] bf16 (transposed)

  const int tid  = threadIdx.x;
  const int wave = tid >> 5;
  const int lane = tid & 31;
  const int half = lane >> 4;     // 0/1: which K-half this lane holds
  const int l16  = lane & 15;

  const int m0 = blockIdx.x * 128;
  const int n0 = blockIdx.y * 128;
  const int mW = (wave & 1) * 64;   // 2 waves along M
  const int nW = (wave >> 1) * 32;  // 4 waves along N

  v8f acc[4][2] = {};

  for (int k0 = 0; k0 < D_; k0 += 32) {
    if (tid < 128) {
      // --- stage one full X row: 32 fp32 -> 32 bf16, 4 x b128 stores ------
      const int m = tid;
      const float* src = X + (size_t)(m0 + m) * D_ + k0;
      unsigned short tmp[32];
#pragma unroll
      for (int v = 0; v < 8; v++) {
        float4 f = *(const float4*)(src + v * 4);
        tmp[v*4+0] = f2bf(f.x); tmp[v*4+1] = f2bf(f.y);
        tmp[v*4+2] = f2bf(f.z); tmp[v*4+3] = f2bf(f.w);
      }
      uint4* dst = (uint4*)(Xs + m * 32);
#pragma unroll
      for (int v = 0; v < 4; v++) dst[v] = ((const uint4*)tmp)[v];
    } else {
      // --- stage W 4k x 8n tile, register transpose, 8 x b64 stores -------
      const int idx = tid - 128;       // 0..127
      const int kg  = idx >> 4;        // 0..7  -> k base kg*4
      const int ng  = idx & 15;        // 0..15 -> n base ng*8
      unsigned short buf[4][8];
#pragma unroll
      for (int kk = 0; kk < 4; kk++) {
        const float* src = W + (size_t)(k0 + kg*4 + kk) * A_ + n0 + ng*8;
        float4 f0 = *(const float4*)(src);
        float4 f1 = *(const float4*)(src + 4);
        buf[kk][0] = f2bf(f0.x); buf[kk][1] = f2bf(f0.y);
        buf[kk][2] = f2bf(f0.z); buf[kk][3] = f2bf(f0.w);
        buf[kk][4] = f2bf(f1.x); buf[kk][5] = f2bf(f1.y);
        buf[kk][6] = f2bf(f1.z); buf[kk][7] = f2bf(f1.w);
      }
#pragma unroll
      for (int nn = 0; nn < 8; nn++) {
        uint2 u;
        u.x = (uint32_t)buf[0][nn] | ((uint32_t)buf[1][nn] << 16);
        u.y = (uint32_t)buf[2][nn] | ((uint32_t)buf[3][nn] << 16);
        *(uint2*)(Ws + (ng*8 + nn) * 32 + kg*4) = u;
      }
    }
    __syncthreads();

    // --- WMMA: 4 (M) x 2 (N) tiles, K=32 ----------------------------------
    v16bf bfr[2];
#pragma unroll
    for (int j = 0; j < 2; j++) {
      const unsigned short* p = Ws + (nW + j*16 + l16) * 32 + half * 16;
      bfr[j] = load_frag16(p);
    }
#pragma unroll
    for (int i = 0; i < 4; i++) {
      const unsigned short* q = Xs + (mW + i*16 + l16) * 32;
      v16bf af = load_frag_pair(q + half * 16, q + half * 16 + 8);
#pragma unroll
      for (int j = 0; j < 2; j++)
        acc[i][j] = wmma_bf16(af, bfr[j], acc[i][j]);
    }
    __syncthreads();
  }

  // --- epilogue ------------------------------------------------------------
  if (blockIdx.z == 2) {
    // V: store transposed. Each lane's 8 accum values (r=0..7) are 8
    // consecutive s-positions for one column n -> one b128 store per tile.
    const int bIdx  = m0 / S_;                       // 128 | 4096: same batch
    const int sBase = (m0 - bIdx * S_) + mW + half * 8;
#pragma unroll
    for (int i = 0; i < 4; i++)
#pragma unroll
      for (int j = 0; j < 2; j++) {
        const int n = n0 + nW + j*16 + l16;
        uint4 u;
        u.x = (uint32_t)f2bf(acc[i][j][0]) | ((uint32_t)f2bf(acc[i][j][1]) << 16);
        u.y = (uint32_t)f2bf(acc[i][j][2]) | ((uint32_t)f2bf(acc[i][j][3]) << 16);
        u.z = (uint32_t)f2bf(acc[i][j][4]) | ((uint32_t)f2bf(acc[i][j][5]) << 16);
        u.w = (uint32_t)f2bf(acc[i][j][6]) | ((uint32_t)f2bf(acc[i][j][7]) << 16);
        *(uint4*)(Out + ((size_t)bIdx * A_ + n) * S_ + sBase + i*16) = u;
      }
  } else {
    // Q, K: row-major scalar bf16 stores
#pragma unroll
    for (int i = 0; i < 4; i++)
#pragma unroll
      for (int j = 0; j < 2; j++)
#pragma unroll
        for (int r = 0; r < 8; r++) {
          const int m = m0 + mW + i*16 + half*8 + r;   // C layout: M = r (+8)
          const int n = n0 + nW + j*16 + l16;          // N = lane%16
          Out[(size_t)m * A_ + n] = f2bf(acc[i][j][r]);
        }
  }
}

// ===========================================================================
// Kernel 2: flash attention over bf16 Q/K/Vt, f32 output (no 1/sqrt(d)).
//   Block = (batch b, 16-query tile). 8 waves, each owns a 128-wide A-slice.
//   K and V tiles both staged with GLOBAL_LOAD_ASYNC_TO_LDS_B128 (ASYNCcnt);
//   V needs no transpose work since kernel 1 wrote Vt[b][a][s].
// ===========================================================================
#define QT   16
#define KT   64
#define LDS_QLS   (QT * A_ * 2)               // 32768
#define LDS_KV    (KT * A_ * 2)               // 131072  (K rows / Vt rows)
#define LDS_SPART (8 * QT * KT * 4)           // 32768
#define LDS_SSUM  (QT * KT * 4)               // 4096
#define LDS_P     (QT * KT * 2)               // 2048
#define LDS_MISC  (2 * QT * 4)                // corr + linv
#define LDS_TOTAL (LDS_QLS + LDS_KV + LDS_SPART + LDS_SSUM + LDS_P + LDS_MISC)

__global__ __launch_bounds__(256) void attn(
    const unsigned short* __restrict__ Qb,
    const unsigned short* __restrict__ Kb,
    const unsigned short* __restrict__ Vt,
    float* __restrict__ out)
{
  extern __shared__ __align__(16) char smem[];   // dynamic LDS, base offset 0
  unsigned short* Qls   = (unsigned short*)(smem);
  unsigned short* KV    = (unsigned short*)(smem + LDS_QLS);
  float*          Spart = (float*)(smem + LDS_QLS + LDS_KV);
  float*          Ssum  = Spart + 8 * QT * KT;
  unsigned short* P     = (unsigned short*)(Ssum + QT * KT);
  float*          corr  = (float*)(P + QT * KT);
  float*          linv  = corr + QT;

  const int tid  = threadIdx.x;
  const int wave = tid >> 5;
  const int lane = tid & 31;
  const int half = lane >> 4;
  const int l16  = lane & 15;

  const int b  = blockIdx.y;
  const int q0 = blockIdx.x * QT;
  const unsigned short* Qg  = Qb + ((size_t)b * S_ + q0) * A_;
  const unsigned short* Kg  = Kb + (size_t)b * S_ * A_;
  const unsigned short* Vtg = Vt + (size_t)b * A_ * S_;   // [a][s]

  // --- load Q tile 16x1024 once -------------------------------------------
  {
    const int q = tid >> 4;            // 0..15
    const int s = (tid & 15) * 64;     // 64 elems per thread
    const uint4* src = (const uint4*)(Qg + (size_t)q * A_ + s);
    uint4* dst = (uint4*)(Qls + q * A_ + s);
#pragma unroll
    for (int v = 0; v < 8; v++) dst[v] = src[v];
  }

  const int aW = wave * 128;           // this wave's A-slice base

  v8f   oacc[8] = {};
  float mrow = -3.0e38f;               // running row max (wave 0, lanes 0..15)
  float lrow = 0.0f;                   // running row sum

  for (int kt = 0; kt < S_; kt += KT) {
    __syncthreads();   // prev P·V done; safe to overwrite KV

    // --- stage K tile 64x1024 row-major via async global->LDS b128 --------
    {
      const int key = tid >> 2;                  // 0..63
      const int s   = (tid & 3) * 256;           // 256 elems per thread
      const unsigned short* src = Kg + (size_t)(kt + key) * A_ + s;
      const unsigned kvBase = LDS_QLS + (unsigned)(key * A_ + s) * 2;
#pragma unroll
      for (int v = 0; v < 32; v++)
        async_g2lds_b128(kvBase + v * 16, src + v * 8);
      // hint: start pulling this iteration's Vt rows toward L2
      const int a0 = tid * 4;
#pragma unroll
      for (int j = 0; j < 4; j++)
        __builtin_prefetch(Vtg + (size_t)(a0 + j) * S_ + kt, 0, 1);
      wait_asynccnt0();
    }
    __syncthreads();

    // --- phase A: per-wave partial scores S[16x64] over its 128 A-slice ---
    {
      v8f sacc[4] = {};
#pragma unroll
      for (int c = 0; c < 4; c++) {
        const int a = aW + c * 32;
        const unsigned short* qp = Qls + l16 * A_ + a;
        v16bf af = load_frag_pair(qp + half * 16, qp + half * 16 + 8);
#pragma unroll
        for (int t = 0; t < 4; t++) {
          // B column n = key (t*16+l16); its reduction values are K's row
          const unsigned short* kp = KV + (t*16 + l16) * A_ + a + half * 16;
          v16bf bf = load_frag16(kp);
          sacc[t] = wmma_bf16(af, bf, sacc[t]);
        }
      }
      float* sp = Spart + wave * (QT * KT);
#pragma unroll
      for (int t = 0; t < 4; t++)
#pragma unroll
        for (int r = 0; r < 8; r++)
          sp[(half*8 + r) * KT + t*16 + l16] = sacc[t][r];
    }
    __syncthreads();

    // --- reduce 8 wave-partials -> Ssum[16x64] ----------------------------
#pragma unroll
    for (int j = 0; j < 4; j++) {
      const int e = tid * 4 + j;       // 256*4 = 1024 elements
      float s = 0.f;
#pragma unroll
      for (int w = 0; w < 8; w++) s += Spart[w * (QT * KT) + e];
      Ssum[e] = s;
    }
    __syncthreads();

    // --- online softmax: wave 0, lane r handles query row r ---------------
    if (wave == 0 && lane < QT) {
      const int r = lane;
      float mnew = mrow;
      for (int c = 0; c < KT; c++) mnew = fmaxf(mnew, Ssum[r * KT + c]);
      const float cfac = __expf(mrow - mnew);
      float sum = 0.f;
      for (int c = 0; c < KT; c++) {
        const float p = __expf(Ssum[r * KT + c] - mnew);
        sum += p;
        P[r * KT + c] = f2bf(p);
      }
      lrow = lrow * cfac + sum;
      mrow = mnew;
      corr[r] = cfac;
    }
    __syncthreads();

    // --- rescale O accumulators; stage Vt tile via async copies -----------
    {
      const int a0 = tid * 4;          // 4 a-rows per thread, 128B each
#pragma unroll
      for (int j = 0; j < 4; j++) {
        const unsigned short* src = Vtg + (size_t)(a0 + j) * S_ + kt;
        const unsigned vBase = LDS_QLS + (unsigned)(a0 + j) * (KT * 2);
#pragma unroll
        for (int v = 0; v < 8; v++)
          async_g2lds_b128(vBase + v * 16, src + v * 8);
      }
      float cr[8];
#pragma unroll
      for (int r = 0; r < 8; r++) cr[r] = corr[half*8 + r];
#pragma unroll
      for (int t = 0; t < 8; t++)
#pragma unroll
        for (int r = 0; r < 8; r++) oacc[t][r] *= cr[r];
      wait_asynccnt0();
    }
    __syncthreads();

    // --- phase D: O += P[16x64] * V[64 x 128-slice] -----------------------
#pragma unroll
    for (int c = 0; c < 2; c++) {                 // key chunks of 32
      const unsigned short* pp = P + l16 * KT + c * 32;
      v16bf af = load_frag_pair(pp + half * 16, pp + half * 16 + 8);
#pragma unroll
      for (int t = 0; t < 8; t++) {
        const unsigned short* vp = KV + (aW + t*16 + l16) * KT + c*32 + half*16;
        v16bf bf = load_frag16(vp);
        oacc[t] = wmma_bf16(af, bf, oacc[t]);
      }
    }
  }

  __syncthreads();
  if (wave == 0 && lane < QT) linv[lane] = 1.0f / lrow;
  __syncthreads();

  // --- store O / l as fp32 -------------------------------------------------
  float li[8];
#pragma unroll
  for (int r = 0; r < 8; r++) li[r] = linv[half*8 + r];
  float* og = out + ((size_t)b * S_ + q0) * A_;
#pragma unroll
  for (int t = 0; t < 8; t++)
#pragma unroll
    for (int r = 0; r < 8; r++) {
      const int q = half*8 + r;
      const int a = aW + t*16 + l16;
      og[(size_t)q * A_ + a] = oacc[t][r] * li[r];
    }
}

// ===========================================================================
// Launch
// ===========================================================================
extern "C" void kernel_launch(void* const* d_in, const int* in_sizes, int n_in,
                              void* d_out, int out_size, void* d_ws, size_t ws_size,
                              hipStream_t stream) {
  (void)in_sizes; (void)n_in; (void)out_size; (void)ws_size;

  const float* X  = (const float*)d_in[0];
  const float* Wq = (const float*)d_in[1];
  const float* Wk = (const float*)d_in[2];
  const float* Wv = (const float*)d_in[3];

  // Workspace: bf16 Q, K (row-major) and Vt (transposed), 32 MB each.
  unsigned short* Qb = (unsigned short*)d_ws;
  unsigned short* Kb = Qb + (size_t)BS_ * A_;
  unsigned short* Vt = Kb + (size_t)BS_ * A_;

  dim3 g1(BS_ / 128, A_ / 128, 3);
  qkv_gemm<<<g1, 256, 0, stream>>>(X, Wq, Wk, Wv, Qb, Kb, Vt);

  dim3 g2(S_ / QT, B_);
  attn<<<g2, 256, LDS_TOTAL, stream>>>(Qb, Kb, Vt, (float*)d_out);
}
